// DTransformerLayer_29222957482714
// MI455X (gfx1250) — matrix operand order repaired and checked by reading, not compiled, not run on hardware
//
#include <hip/hip_runtime.h>
#include <hip/hip_fp16.h>
#include <math.h>

// Problem constants (B,S,D,H) = (4,1024,1024,16), DK=64, K_INDEX=5
#define Bn   4
#define Sn   1024
#define Dn   1024
#define Hn   16
#define DKn  64
#define KIDX 5
#define NEGF (-3.402823466e38f)

typedef _Float16 v16h __attribute__((ext_vector_type(16)));
typedef float    v8f  __attribute__((ext_vector_type(8)));

union V16H { v16h v; float4 f[2]; _Float16 h[16]; };

// D = A(16x32 f16) x B(32x16 f16) + C(16x16 f32)  -> v_wmma_f32_16x16x32_f16
__device__ __forceinline__ v8f wmma16(v16h a, v16h b, v8f c) {
  return __builtin_amdgcn_wmma_f32_16x16x32_f16(false, a, false, b, (short)0, c,
                                                false, false);
}

// wave32 reductions / scan (warpSize == 32 on gfx1250)
__device__ __forceinline__ float wave_max(float x) {
#pragma unroll
  for (int m = 16; m > 0; m >>= 1) x = fmaxf(x, __shfl_xor(x, m, 32));
  return x;
}
__device__ __forceinline__ float wave_sum(float x) {
#pragma unroll
  for (int m = 16; m > 0; m >>= 1) x += __shfl_xor(x, m, 32);
  return x;
}

// ---------------------------------------------------------------------------
// K0: f32 -> f16 weight conversion
// ---------------------------------------------------------------------------
__global__ __launch_bounds__(256) void cvt_f32_f16(const float* __restrict__ x,
                                                   _Float16* __restrict__ y, int n) {
  int i = blockIdx.x * 256 + threadIdx.x;
  if (i < n) y[i] = (_Float16)x[i];
}

// ---------------------------------------------------------------------------
// K1: Y(f16) = X(f32) @ W^T(f16) + bias.   One wave per 16x16 output tile.
// W is row-major [n][k] so the B-fragment (lane = column n of B = row n of W)
// reads 16 contiguous K per lane -> two b128 loads.
// ---------------------------------------------------------------------------
__global__ __launch_bounds__(32) void gemm_proj(const float* __restrict__ X,
                                                const _Float16* __restrict__ W,
                                                const float* __restrict__ bias,
                                                _Float16* __restrict__ Y) {
  int lane = threadIdx.x & 31;
  int nn = lane & 15, hf = lane >> 4;
  int n0 = blockIdx.x * 16, m0 = blockIdx.y * 16;
  const float*    xr = X + (size_t)(m0 + nn) * Dn;  // A: lane holds row m=nn
  const _Float16* wr = W + (size_t)(n0 + nn) * Dn;  // B: lane holds col n=nn
  v8f acc = {};
  for (int k0 = 0; k0 < Dn; k0 += 32) {
    __builtin_prefetch(xr + k0 + 128, 0, 0);         // global_prefetch_b8
    __builtin_prefetch(wr + k0 + 256, 0, 0);
    V16H aU, bU;
#pragma unroll
    for (int e = 0; e < 8; ++e) {  // A frag: k = k0 + {hf*16+e, hf*16+8+e}
      aU.h[e]     = (_Float16)xr[k0 + hf * 16 + e];
      aU.h[e + 8] = (_Float16)xr[k0 + hf * 16 + 8 + e];
    }
    const _Float16* wp = wr + k0 + hf * 16;          // B frag: k = k0+hf*16+e
    bU.f[0] = *(const float4*)(wp);
    bU.f[1] = *(const float4*)(wp + 8);
    acc = wmma16(aU.v, bU.v, acc);
  }
#pragma unroll
  for (int r = 0; r < 8; ++r) {                      // C: m = r + hf*8, n = nn
    int m = m0 + r + hf * 8, n = n0 + nn;
    Y[(size_t)m * Dn + n] = (_Float16)(acc[r] + bias[n]);
  }
}

// ---------------------------------------------------------------------------
// K1T: same GEMM, but the epilogue stores V in per-head TRANSPOSED layout
// vT[b][h][dk][s]  so the p@V WMMA B-fragment reads are contiguous b128s.
// ---------------------------------------------------------------------------
__global__ __launch_bounds__(32) void gemm_projT(const float* __restrict__ X,
                                                 const _Float16* __restrict__ W,
                                                 const float* __restrict__ bias,
                                                 _Float16* __restrict__ vT) {
  int lane = threadIdx.x & 31;
  int nn = lane & 15, hf = lane >> 4;
  int n0 = blockIdx.x * 16, m0 = blockIdx.y * 16;
  const float*    xr = X + (size_t)(m0 + nn) * Dn;
  const _Float16* wr = W + (size_t)(n0 + nn) * Dn;
  v8f acc = {};
  for (int k0 = 0; k0 < Dn; k0 += 32) {
    __builtin_prefetch(xr + k0 + 128, 0, 0);
    __builtin_prefetch(wr + k0 + 256, 0, 0);
    V16H aU, bU;
#pragma unroll
    for (int e = 0; e < 8; ++e) {
      aU.h[e]     = (_Float16)xr[k0 + hf * 16 + e];
      aU.h[e + 8] = (_Float16)xr[k0 + hf * 16 + 8 + e];
    }
    const _Float16* wp = wr + k0 + hf * 16;
    bU.f[0] = *(const float4*)(wp);
    bU.f[1] = *(const float4*)(wp + 8);
    acc = wmma16(aU.v, bU.v, acc);
  }
#pragma unroll
  for (int r = 0; r < 8; ++r) {
    int m = m0 + r + hf * 8, n = n0 + nn;
    int bb = m >> 10, j = m & (Sn - 1);   // token -> (batch, seq)
    int hh = n >> 6,  dk = n & (DKn - 1); // feature -> (head, dk)
    vT[(((size_t)bb * Hn + hh) * DKn + dk) * Sn + j] =
        (_Float16)(acc[r] + bias[n]);
  }
}

// ---------------------------------------------------------------------------
// K5a: Y(f32) = A(f16) @ W^T(f16) + bias + residual   (output projection)
// ---------------------------------------------------------------------------
__global__ __launch_bounds__(32) void gemm_out(const _Float16* __restrict__ A,
                                               const _Float16* __restrict__ W,
                                               const float* __restrict__ bias,
                                               const float* __restrict__ resid,
                                               float* __restrict__ Y) {
  int lane = threadIdx.x & 31;
  int nn = lane & 15, hf = lane >> 4;
  int n0 = blockIdx.x * 16, m0 = blockIdx.y * 16;
  const _Float16* ar = A + (size_t)(m0 + nn) * Dn;
  const _Float16* wr = W + (size_t)(n0 + nn) * Dn;
  v8f acc = {};
  for (int k0 = 0; k0 < Dn; k0 += 32) {
    __builtin_prefetch(ar + k0 + 256, 0, 0);
    __builtin_prefetch(wr + k0 + 256, 0, 0);
    V16H aU, bU;
    aU.f[0] = *(const float4*)(ar + k0 + hf * 16);
    aU.f[1] = *(const float4*)(ar + k0 + hf * 16 + 8);
    const _Float16* wp = wr + k0 + hf * 16;
    bU.f[0] = *(const float4*)(wp);
    bU.f[1] = *(const float4*)(wp + 8);
    acc = wmma16(aU.v, bU.v, acc);
  }
#pragma unroll
  for (int r = 0; r < 8; ++r) {
    int m = m0 + r + hf * 8, n = n0 + nn;
    Y[(size_t)m * Dn + n] = acc[r] + bias[n] + resid[(size_t)m * Dn + n];
  }
}

// ---------------------------------------------------------------------------
// K2: fused attention per (b, h, 16-row query tile).
// Phase A: scores(16x1024) via WMMA -> LDS (f32, 64 KB).
// Phase B: per-row masked softmax, cumsum, distance decay, 2nd softmax,
//          max-rescale -> p written to d_out sparse region + back into LDS.
// Phase C: attn(16x64) = p @ V via WMMA (V in transposed layout -> b128),
//          stored f16 in concat layout.
// ---------------------------------------------------------------------------
__global__ __launch_bounds__(256) void attn_kernel(const _Float16* __restrict__ qf,
                                                   const _Float16* __restrict__ kf,
                                                   const _Float16* __restrict__ vT,
                                                   const float* __restrict__ gam,
                                                   float* __restrict__ p_out,
                                                   _Float16* __restrict__ attn16) {
  __shared__ float sS[16][Sn];  // 64 KB of the 320 KB WGP LDS
  int itile = blockIdx.x, h = blockIdx.y, b = blockIdx.z;
  int w = threadIdx.x >> 5, lane = threadIdx.x & 31;
  int nn = lane & 15, hf = lane >> 4;

  // ---- Phase A: raw scores = (q . k) / sqrt(DK) ----
  const _Float16* qp = qf + ((size_t)(b * Sn + itile * 16 + nn)) * Dn + h * DKn;
  V16H a0, a1;
  a0.f[0] = *(const float4*)(qp + hf * 16);
  a0.f[1] = *(const float4*)(qp + hf * 16 + 8);
  a1.f[0] = *(const float4*)(qp + 32 + hf * 16);
  a1.f[1] = *(const float4*)(qp + 32 + hf * 16 + 8);
  for (int jt = w; jt < Sn / 16; jt += 8) {
    int j0 = jt * 16;
    const _Float16* kp = kf + ((size_t)(b * Sn + j0 + nn)) * Dn + h * DKn;
    V16H b0, b1;  // B[k=dk][n=j] = K[j][dk] : contiguous dk per lane
    b0.f[0] = *(const float4*)(kp + hf * 16);
    b0.f[1] = *(const float4*)(kp + hf * 16 + 8);
    b1.f[0] = *(const float4*)(kp + 32 + hf * 16);
    b1.f[1] = *(const float4*)(kp + 32 + hf * 16 + 8);
    v8f c = {};
    c = wmma16(a0.v, b0.v, c);
    c = wmma16(a1.v, b1.v, c);
#pragma unroll
    for (int r = 0; r < 8; ++r) sS[r + hf * 8][j0 + nn] = c[r] * 0.125f;
  }
  __syncthreads();

  // ---- Phase B: per-row pipeline (wave w owns rows 2w, 2w+1) ----
  float g = -fabsf(gam[h]);
#pragma unroll 1
  for (int rr = 0; rr < 2; ++rr) {
    int m = w * 2 + rr;
    int qi = itile * 16 + m;
    float x[32], sv[32];
#pragma unroll
    for (int t = 0; t < 32; ++t) x[t] = sS[m][lane * 32 + t];
    // softmax #1 over causally-masked scores (mask: j < qi)
    float mx = NEGF;
#pragma unroll
    for (int t = 0; t < 32; ++t)
      mx = fmaxf(mx, (lane * 32 + t < qi) ? x[t] : NEGF);
    mx = wave_max(mx);
    float lsum = 0.f;
#pragma unroll
    for (int t = 0; t < 32; ++t) {
      float vm = (lane * 32 + t < qi) ? x[t] : NEGF;
      sv[t] = expf(vm - mx);
      lsum += sv[t];
    }
    float sum1 = wave_sum(lsum);
    // inclusive cumsum of sv across the 1024-wide row (blocked per lane)
    float run = 0.f;
#pragma unroll
    for (int t = 0; t < 32; ++t) { run += sv[t]; sv[t] = run; }
    float pre = run;
#pragma unroll
    for (int d2 = 1; d2 < 32; d2 <<= 1) {
      float o = __shfl_up(pre, d2, 32);
      if (lane >= d2) pre += o;
    }
    float off = pre - run;  // exclusive lane offset
    // distance-decay effect applied to the RAW scores, then re-mask
#pragma unroll
    for (int t = 0; t < 32; ++t) {
      int j = lane * 32 + t;
      float rem  = (sum1 - (sv[t] + off)) / sum1;  // disttotal - distcum
      float pe   = fabsf((float)(qi - j));
      float dist = sqrtf(fmaxf(rem * pe, 0.f));
      float eff  = fminf(fmaxf(expf(dist * g), 1e-5f), 1e5f);
      x[t] = (j < qi) ? x[t] * eff : NEGF;
    }
    // softmax #2, zero masked, rescale by min(1/max, 5)
    float mx2 = NEGF;
#pragma unroll
    for (int t = 0; t < 32; ++t) mx2 = fmaxf(mx2, x[t]);
    mx2 = wave_max(mx2);
    float lsum2 = 0.f;
#pragma unroll
    for (int t = 0; t < 32; ++t) { sv[t] = expf(x[t] - mx2); lsum2 += sv[t]; }
    float sum2 = wave_sum(lsum2);
    float pmax = 0.f;
#pragma unroll
    for (int t = 0; t < 32; ++t) {
      int j = lane * 32 + t;
      float p = (j < qi) ? sv[t] / sum2 : 0.f;
      sv[t] = p;
      pmax = fmaxf(pmax, p);
    }
    pmax = wave_max(pmax);
    float scale = fminf(1.0f / pmax, 5.0f);  // pmax==0 -> inf -> clamp to 5
    size_t prow = (((size_t)(b * Hn + h)) * Sn + qi) * (size_t)Sn;
#pragma unroll
    for (int t = 0; t < 32; ++t) {
      int j = lane * 32 + t;
      float p = sv[t] * scale;
      p_out[prow + j] = p;  // consumed later by the sparse top-k kernel
      sS[m][j] = p;         // consumed below by p @ V
    }
  }
  __syncthreads();

  // ---- Phase C: attn = p(16x1024) @ V(1024x64); waves 0..3 own dk-tiles ----
  if (w < 4) {
    int n = w * 16 + nn;  // dk column
    const _Float16* vrow = vT + (((size_t)b * Hn + h) * DKn + n) * Sn;
    v8f c = {};
    for (int ks = 0; ks < Sn / 32; ++ks) {
      V16H aU, bU;
#pragma unroll
      for (int e = 0; e < 8; ++e) {
        aU.h[e]     = (_Float16)sS[nn][ks * 32 + hf * 16 + e];
        aU.h[e + 8] = (_Float16)sS[nn][ks * 32 + hf * 16 + 8 + e];
      }
      // B[k=j][n=dk] = V[j][dk] = vT[dk][j] : 16 contiguous j -> two b128s
      const _Float16* vp = vrow + ks * 32 + hf * 16;
      bU.f[0] = *(const float4*)(vp);
      bU.f[1] = *(const float4*)(vp + 8);
      c = wmma16(aU.v, bU.v, c);
    }
#pragma unroll
    for (int r = 0; r < 8; ++r) {  // store f16 in concat [b, i, h*DK+dk] layout
      int m = r + hf * 8;
      attn16[((size_t)(b * Sn + itile * 16 + m)) * Dn + h * DKn + n] =
          (_Float16)c[r];
    }
  }
}

// ---------------------------------------------------------------------------
// K4: in-place sparse top-5 softmax on p rows. One wave per 1024-wide row.
// Rows i < 5 : plain softmax. Rows i >= 5 : keep values >= 5th-largest
// (multiset order statistic, duplicate-safe), others -> NEG, then softmax.
// ---------------------------------------------------------------------------
__global__ __launch_bounds__(256) void sparse_kernel(float* __restrict__ p) {
  int row = blockIdx.x * 8 + (threadIdx.x >> 5);
  int lane = threadIdx.x & 31;
  int i = row & (Sn - 1);  // query index within [B,H,S]
  float* pr = p + (size_t)row * Sn;
  float v[32];
#pragma unroll
  for (int t = 0; t < 32; ++t) v[t] = pr[lane * 32 + t];
  if (i >= KIDX) {
    unsigned consumed = 0;
    float kth = 0.f;
#pragma unroll 1
    for (int it = 0; it < KIDX; ++it) {
      float lmax = NEGF;
      int li = -1;
#pragma unroll
      for (int t = 0; t < 32; ++t)
        if (!((consumed >> t) & 1u) && v[t] > lmax) { lmax = v[t]; li = t; }
      float gmax = wave_max(lmax);
      unsigned long long bal = __ballot((lmax == gmax) && (li >= 0));
      int fl = __ffsll((long long)bal) - 1;
      if (lane == fl) consumed |= (1u << li);  // consume one instance
      kth = gmax;
    }
#pragma unroll
    for (int t = 0; t < 32; ++t)
      if (v[t] < kth) v[t] = NEGF;
  }
  float mx = NEGF;
#pragma unroll
  for (int t = 0; t < 32; ++t) mx = fmaxf(mx, v[t]);
  mx = wave_max(mx);
  float ls = 0.f;
#pragma unroll
  for (int t = 0; t < 32; ++t) { v[t] = expf(v[t] - mx); ls += v[t]; }
  float s = wave_sum(ls);
#pragma unroll
  for (int t = 0; t < 32; ++t) pr[lane * 32 + t] = v[t] / s;
}

// ---------------------------------------------------------------------------
// K5b: in-place LayerNorm over D=1024. One wave per row.
// ---------------------------------------------------------------------------
__global__ __launch_bounds__(256) void ln_kernel(float* __restrict__ y,
                                                 const float* __restrict__ lw,
                                                 const float* __restrict__ lb) {
  int row = blockIdx.x * 8 + (threadIdx.x >> 5);
  int lane = threadIdx.x & 31;
  float* yr = y + (size_t)row * Dn;
  float v[32];
  float s = 0.f, sq = 0.f;
#pragma unroll
  for (int t = 0; t < 32; ++t) {
    v[t] = yr[lane * 32 + t];
    s += v[t];
    sq += v[t] * v[t];
  }
  s = wave_sum(s);
  sq = wave_sum(sq);
  float mu = s * (1.0f / Dn);
  float var = sq * (1.0f / Dn) - mu * mu;
  float inv = 1.0f / sqrtf(var + 1e-5f);
#pragma unroll
  for (int t = 0; t < 32; ++t) {
    int col = lane * 32 + t;
    yr[col] = (v[t] - mu) * inv * lw[col] + lb[col];
  }
}

// ---------------------------------------------------------------------------
extern "C" void kernel_launch(void* const* d_in, const int* in_sizes, int n_in,
                              void* d_out, int out_size, void* d_ws, size_t ws_size,
                              hipStream_t stream) {
  (void)in_sizes; (void)n_in; (void)out_size; (void)ws_size;
  const float* query  = (const float*)d_in[0];
  const float* key    = (const float*)d_in[1];
  const float* values = (const float*)d_in[2];
  /* d_in[3] = lens : unused by the reference */
  const float* Wq     = (const float*)d_in[4];
  const float* bq     = (const float*)d_in[5];
  const float* Wv     = (const float*)d_in[6];
  const float* bv     = (const float*)d_in[7];
  const float* Wo     = (const float*)d_in[8];
  const float* bo     = (const float*)d_in[9];
  const float* gammas = (const float*)d_in[10];
  const float* ln_w   = (const float*)d_in[11];
  const float* ln_b   = (const float*)d_in[12];

  const size_t BSD = (size_t)Bn * Sn * Dn;  // 4,194,304
  const size_t DD  = (size_t)Dn * Dn;       // 1,048,576
  _Float16* ws     = (_Float16*)d_ws;       // ~40 MB of f16 scratch
  _Float16* qf     = ws;
  _Float16* kf     = ws + BSD;
  _Float16* vT     = ws + 2 * BSD;          // [B,H,DK,S] transposed V
  _Float16* wq16   = ws + 3 * BSD;
  _Float16* wv16   = wq16 + DD;
  _Float16* wo16   = wv16 + DD;
  _Float16* attn16 = wo16 + DD;

  float* outp  = (float*)d_out;  // [B,S,D] f32
  float* p_out = outp + BSD;     // [B,H,S,S] f32 (sparse_scores region)

  int cvb = (int)((DD + 255) / 256);
  cvt_f32_f16<<<cvb, 256, 0, stream>>>(Wq, wq16, (int)DD);
  cvt_f32_f16<<<cvb, 256, 0, stream>>>(Wv, wv16, (int)DD);
  cvt_f32_f16<<<cvb, 256, 0, stream>>>(Wo, wo16, (int)DD);

  dim3 gg(Dn / 16, (Bn * Sn) / 16);  // 64 x 256 waves
  gemm_proj <<<gg, 32, 0, stream>>>(query,  wq16, bq, qf);
  gemm_proj <<<gg, 32, 0, stream>>>(key,    wq16, bq, kf);
  gemm_projT<<<gg, 32, 0, stream>>>(values, wv16, bv, vT);

  attn_kernel<<<dim3(Sn / 16, Hn, Bn), 256, 0, stream>>>(qf, kf, vT, gammas,
                                                         p_out, attn16);

  sparse_kernel<<<dim3((Bn * Hn * Sn) / 8), 256, 0, stream>>>(p_out);

  gemm_out<<<gg, 32, 0, stream>>>(attn16, wo16, bo, query, outp);
  ln_kernel<<<dim3((Bn * Sn) / 8), 256, 0, stream>>>(outp, ln_w, ln_b);
}
